// MalwareGCN_72232759984522
// MI455X (gfx1250) — matrix operand order repaired and checked by reading, not compiled, not run on hardware
//
#include <hip/hip_runtime.h>
#include <hip/hip_bf16.h>
#include <math.h>

// ---------------- problem constants (match reference) ----------------
#define N_NODES   20000
#define N_EDGES   300000
#define N_GRAPH   64
#define INCH      64
#define EDGE_DIM  7
#define NEG_SLOPE 0.2f
#define BN_EPS    1e-5f

typedef _Float16 half_t;
typedef half_t v16h __attribute__((ext_vector_type(16)));
typedef half_t v8h  __attribute__((ext_vector_type(8)));
typedef float  v8f  __attribute__((ext_vector_type(8)));

// float atomic max via signed/unsigned integer ordering trick
__device__ __forceinline__ void atomicMaxFloat(float* addr, float val) {
  if (val >= 0.0f) atomicMax(reinterpret_cast<int*>(addr), __float_as_int(val));
  else             atomicMin(reinterpret_cast<unsigned int*>(addr), __float_as_uint(val));
}

// ---------------- utility kernels ----------------
__global__ void fill_f32_kernel(float* __restrict__ p, float v, int n) {
  int i = blockIdx.x * blockDim.x + threadIdx.x;
  if (i < n) p[i] = v;
}

__global__ void cvt_f32_f16_kernel(const float* __restrict__ s, half_t* __restrict__ d, int n) {
  int i = blockIdx.x * blockDim.x + threadIdx.x;
  if (i < n) d[i] = (half_t)s[i];
}

// W[K,M] f32 -> WT[M,K] f16 (transposed weights: makes WMMA B-fragments contiguous)
__global__ void cvt_transpose_f16_kernel(const float* __restrict__ W, half_t* __restrict__ WT,
                                         int K, int M) {
  int i = blockIdx.x * blockDim.x + threadIdx.x;
  if (i >= K * M) return;
  int k = i / M, m = i - k * M;
  WT[(size_t)m * K + k] = (half_t)W[i];
}

// degree + per-destination edge_attr sum (for self-loop fill_value='mean')
__global__ void deg_attr_kernel(const int* __restrict__ dst, const float* __restrict__ eattr,
                                float* __restrict__ deg, float* __restrict__ lattr, int e) {
  int i = blockIdx.x * blockDim.x + threadIdx.x;
  if (i >= e) return;
  int d = dst[i];
  atomicAdd(&deg[d], 1.0f);
#pragma unroll
  for (int k = 0; k < EDGE_DIM; ++k)
    atomicAdd(&lattr[d * EDGE_DIM + k], eattr[i * EDGE_DIM + k]);
}

__global__ void loop_div_kernel(float* __restrict__ lattr, const float* __restrict__ deg, int n) {
  int i = blockIdx.x * blockDim.x + threadIdx.x;
  if (i >= n) return;
  float inv = 1.0f / fmaxf(deg[i], 1.0f);
#pragma unroll
  for (int k = 0; k < EDGE_DIM; ++k) lattr[i * EDGE_DIM + k] *= inv;
}

// ---------------- WMMA GEMM ----------------
// C[M,N] = A[M,K](f16, row-major) x B[K,N] where B is supplied TRANSPOSED as BT[N,K] (f16).
// Block: 128 threads = 4 waves. Block tile: 32 rows x 64 cols.
//  - A tile (32x32 halves per K-step) staged into LDS with global_load_async_to_lds_b128,
//    shared by all 4 waves (CDNA5 async path, ASYNCcnt + barrier).
//  - Each wave: one 16-col tile, two 16-row sub-tiles -> 2 v_wmma per K-step, B frag reused.
// LDS row stride 40 halves (80B = 5*16B): v8h reads stay 16B-aligned, no bank conflicts.
__global__ __launch_bounds__(128)
void gemm_wmma_f16(const half_t* __restrict__ A, const half_t* __restrict__ BT,
                   float* __restrict__ C, int K, int N) {
  __shared__ half_t As[32 * 40];
  const int tid  = threadIdx.x;
  const int wave = tid >> 5;
  const int lane = tid & 31;
  const int tm = blockIdx.x;                 // 32-row tile (M/32 exact)
  const int tn = blockIdx.y * 4 + wave;      // 16-col tile per wave
  const bool active = (tn * 16) < N;         // wave-uniform
  const int col = active ? (tn * 16 + (lane & 15)) : (N - 1);  // clamp loads for idle waves
  const int hiA = (lane >> 4) ? 8 : 0;       // A frag: lanes>=16 hold K {8..15,24..31}
  const int hiB = (lane >> 4) ? 8 : 0;       // B frag: lanes>=16 hold K {8..15,24..31}

  // staging assignment: thread t -> row t/4, 8-half (16B) segment t%4
  const int srow = tid >> 2;
  const int sseg = (tid & 3) * 8;
  const half_t* astage = A + (size_t)(tm * 32 + srow) * K + sseg;
  const unsigned ldsoff = (unsigned)(size_t)&As[srow * 40 + sseg]; // LDS byte offset (low 32 bits)

  const half_t* bcol = BT + (size_t)col * K;
  const half_t* a0p = &As[(lane & 15) * 40 + hiA];
  const half_t* a1p = &As[((lane & 15) + 16) * 40 + hiA];

  v8f acc0 = {}, acc1 = {};
  for (int kk = 0; kk < K; kk += 32) {
    // async stage the 32x32-half A tile into LDS (tracked by ASYNCcnt)
    asm volatile("global_load_async_to_lds_b128 %0, %1, off"
                 :: "v"(ldsoff), "v"(astage + kk) : "memory");
    asm volatile("s_wait_asynccnt 0" ::: "memory");
    __syncthreads();

    __builtin_prefetch(bcol + kk + 32, 0, 0);             // next B chunk -> global_prefetch_b8
    v8h bl = *(const v8h*)(bcol + kk + hiB);
    v8h bh = *(const v8h*)(bcol + kk + hiB + 16);
    v16h bf = __builtin_shufflevector(bl, bh, 0,1,2,3,4,5,6,7,8,9,10,11,12,13,14,15);

    v8h a0l = *(const v8h*)(a0p);
    v8h a0h = *(const v8h*)(a0p + 16);
    v8h a1l = *(const v8h*)(a1p);
    v8h a1h = *(const v8h*)(a1p + 16);
    v16h af0 = __builtin_shufflevector(a0l, a0h, 0,1,2,3,4,5,6,7,8,9,10,11,12,13,14,15);
    v16h af1 = __builtin_shufflevector(a1l, a1h, 0,1,2,3,4,5,6,7,8,9,10,11,12,13,14,15);

    acc0 = __builtin_amdgcn_wmma_f32_16x16x32_f16(false, af0, false, bf, (short)0, acc0, false, false);
    acc1 = __builtin_amdgcn_wmma_f32_16x16x32_f16(false, af1, false, bf, (short)0, acc1, false, false);
    __syncthreads();
  }
  if (active) {
    // C/D layout: VGPR v -> M = v + 8*(lane>=16), N = lane&15
    const int cc = tn * 16 + (lane & 15);
    const int r0 = tm * 32 + (lane >> 4) * 8;
#pragma unroll
    for (int v = 0; v < 8; ++v) C[(size_t)(r0 + v) * N + cc] = acc0[v];
#pragma unroll
    for (int v = 0; v < 8; ++v) C[(size_t)(r0 + 16 + v) * N + cc] = acc1[v];
  }
}

// ---------------- attention pieces ----------------
__global__ void attn_node_kernel(const float* __restrict__ H, const float* __restrict__ a_s,
                                 const float* __restrict__ a_d, float* __restrict__ asrc,
                                 float* __restrict__ adst, int n, int heads, int C) {
  int i = blockIdx.x * blockDim.x + threadIdx.x;
  if (i >= n * heads) return;
  int node = i / heads, h = i - node * heads;
  const float* hp = H + (size_t)node * heads * C + h * C;
  const float* aw = a_s + h * C;
  const float* dw = a_d + h * C;
  float s0 = 0.f, s1 = 0.f;
  for (int c = 0; c < C; ++c) { float v = hp[c]; s0 += v * aw[c]; s1 += v * dw[c]; }
  asrc[i] = s0; adst[i] = s1;
}

// Ve[k,h] = sum_c We[k, h*C+c] * a_e[h,c]  (folds the edge GEMM into a 7xH GEMV)
__global__ void ve_kernel(const float* __restrict__ We, const float* __restrict__ a_e,
                          float* __restrict__ Ve, int heads, int C) {
  int i = threadIdx.x;
  if (i >= EDGE_DIM * heads) return;
  int k = i / heads, h = i - k * heads;
  int M = heads * C;
  float s = 0.f;
  for (int c = 0; c < C; ++c) s += We[k * M + h * C + c] * a_e[h * C + c];
  Ve[k * heads + h] = s;
}

__global__ void aedge_kernel(const float* __restrict__ eattr, const float* __restrict__ lattr,
                             const float* __restrict__ Ve, float* __restrict__ aedge, int heads) {
  int e = blockIdx.x * blockDim.x + threadIdx.x;
  const int Etot = N_EDGES + N_NODES;
  if (e >= Etot) return;
  const float* p = (e < N_EDGES) ? (eattr + (size_t)e * EDGE_DIM)
                                 : (lattr + (size_t)(e - N_EDGES) * EDGE_DIM);
  float v[EDGE_DIM];
#pragma unroll
  for (int k = 0; k < EDGE_DIM; ++k) v[k] = p[k];
  for (int h = 0; h < heads; ++h) {
    float s = 0.f;
#pragma unroll
    for (int k = 0; k < EDGE_DIM; ++k) s += v[k] * Ve[k * heads + h];
    aedge[e * heads + h] = s;
  }
}

__global__ void edge_logit_kernel(const int* __restrict__ src, const int* __restrict__ dst,
                                  const float* __restrict__ asrc, const float* __restrict__ adst,
                                  const float* __restrict__ aedge, float* __restrict__ pbuf,
                                  float* __restrict__ mbuf, int H) {
  const int Etot = N_EDGES + N_NODES;
  int idx = blockIdx.x * blockDim.x + threadIdx.x;
  if (idx >= Etot * H) return;
  int e = idx / H, h = idx - e * H;
  int s, d;
  if (e < N_EDGES) { s = src[e]; d = dst[e]; } else { s = d = e - N_EDGES; }
  float l = asrc[s * H + h] + adst[d * H + h] + aedge[idx];
  l = (l > 0.f) ? l : NEG_SLOPE * l;
  pbuf[idx] = l;
  atomicMaxFloat(&mbuf[d * H + h], l);
}

__global__ void edge_exp_kernel(const int* __restrict__ dst, float* __restrict__ pbuf,
                                const float* __restrict__ mbuf, float* __restrict__ dbuf, int H) {
  const int Etot = N_EDGES + N_NODES;
  int idx = blockIdx.x * blockDim.x + threadIdx.x;
  if (idx >= Etot * H) return;
  int e = idx / H, h = idx - e * H;
  int d = (e < N_EDGES) ? dst[e] : (e - N_EDGES);
  float p = __expf(pbuf[idx] - mbuf[d * H + h]);
  pbuf[idx] = p;
  atomicAdd(&dbuf[d * H + h], p);
}

__global__ void edge_msg_kernel(const int* __restrict__ src, const int* __restrict__ dst,
                                const float* __restrict__ pbuf, const float* __restrict__ dbuf,
                                const float* __restrict__ H, float* __restrict__ out,
                                int heads, int C) {
  const int Etot = N_EDGES + N_NODES;
  const int M = heads * C;
  long long idx = (long long)blockIdx.x * blockDim.x + threadIdx.x;
  if (idx >= (long long)Etot * M) return;
  int e = (int)(idx / M);
  int j = (int)(idx - (long long)e * M);
  int h = j / C;
  int s, d;
  if (e < N_EDGES) { s = src[e]; d = dst[e]; } else { s = d = e - N_EDGES; }
  float alpha = pbuf[e * heads + h] / dbuf[d * heads + h];
  atomicAdd(&out[(size_t)d * M + j], alpha * H[(size_t)s * M + j]);
}

__global__ void post_bn_kernel(const float* __restrict__ acc, const float* __restrict__ b,
                               const float* __restrict__ g, const float* __restrict__ bt,
                               const float* __restrict__ rm, const float* __restrict__ rv,
                               float* __restrict__ Y, half_t* __restrict__ Y16, int n, int M) {
  int i = blockIdx.x * blockDim.x + threadIdx.x;
  if (i >= n * M) return;
  int j = i % M;
  float v = acc[i] + b[j];
  v = (v - rm[j]) * (g[j] * rsqrtf(rv[j] + BN_EPS)) + bt[j];
  v = (v > 0.f) ? v : 0.f;
  Y[i] = v;
  Y16[i] = (half_t)v;
}

// ---------------- pooling + MLP head ----------------
__global__ void pool_kernel(const float* __restrict__ h3, const int* __restrict__ batch,
                            float* __restrict__ psum, float* __restrict__ pmax,
                            float* __restrict__ pcnt, int n) {
  int i = blockIdx.x * blockDim.x + threadIdx.x;
  if (i >= n) return;
  int g = batch[i];
  atomicAdd(&pcnt[g], 1.0f);
#pragma unroll
  for (int c = 0; c < 32; ++c) {
    float v = h3[(size_t)i * 32 + c];
    atomicAdd(&psum[g * 32 + c], v);
    atomicMaxFloat(&pmax[g * 32 + c], v);
  }
}

__global__ void mlp_kernel(const float* __restrict__ psum, const float* __restrict__ pmax,
                           const float* __restrict__ pcnt, const float* __restrict__ lw1,
                           const float* __restrict__ lb1, const float* __restrict__ lw2,
                           const float* __restrict__ lb2, float* __restrict__ out) {
  int g = threadIdx.x;
  if (g >= N_GRAPH) return;
  float z[64];
  float cnt = pcnt[g];
  float inv = 1.0f / fmaxf(cnt, 1.0f);
#pragma unroll
  for (int c = 0; c < 32; ++c) z[c] = psum[g * 32 + c] * inv;
#pragma unroll
  for (int c = 0; c < 32; ++c) z[32 + c] = (cnt > 0.f) ? pmax[g * 32 + c] : 0.f;
  float o = lb2[0];
  for (int j = 0; j < 32; ++j) {
    float a = lb1[j];
    for (int i = 0; i < 64; ++i) a += z[i] * lw1[i * 32 + j];
    a = (a > 0.f) ? a : 0.f;
    o += a * lw2[j];
  }
  out[g] = 1.0f / (1.0f + __expf(-o));
}

// ---------------- orchestration ----------------
extern "C" void kernel_launch(void* const* d_in, const int* in_sizes, int n_in,
                              void* d_out, int out_size, void* d_ws, size_t ws_size,
                              hipStream_t stream) {
  (void)in_sizes; (void)n_in; (void)out_size; (void)ws_size;
  const float* x     = (const float*)d_in[0];
  const int*   ei    = (const int*)d_in[1];
  const float* eattr = (const float*)d_in[2];
  const int*   batch = (const int*)d_in[3];
  const int* src = ei;
  const int* dst = ei + N_EDGES;

  struct LayerP { const float *W, *as_, *ad_, *We, *ae_, *b, *g, *bt, *rm, *rv; int Kin, C, Hh; };
  LayerP L[3];
  for (int li = 0; li < 3; ++li) {
    int base = 4 + li * 10;
    L[li].W  = (const float*)d_in[base + 0];
    L[li].as_= (const float*)d_in[base + 1];
    L[li].ad_= (const float*)d_in[base + 2];
    L[li].We = (const float*)d_in[base + 3];
    L[li].ae_= (const float*)d_in[base + 4];
    L[li].b  = (const float*)d_in[base + 5];
    L[li].g  = (const float*)d_in[base + 6];
    L[li].bt = (const float*)d_in[base + 7];
    L[li].rm = (const float*)d_in[base + 8];
    L[li].rv = (const float*)d_in[base + 9];
  }
  L[0].Kin = INCH; L[0].C = 128; L[0].Hh = 4;   // -> 512
  L[1].Kin = 512;  L[1].C = 64;  L[1].Hh = 4;   // -> 256
  L[2].Kin = 256;  L[2].C = 32;  L[2].Hh = 1;   // -> 32
  const float* lw1 = (const float*)d_in[34];
  const float* lb1 = (const float*)d_in[35];
  const float* lw2 = (const float*)d_in[36];
  const float* lb2 = (const float*)d_in[37];
  float* outp = (float*)d_out;

  // ---- carve workspace ----
  char* base = (char*)d_ws;
  size_t off = 0;
  auto carve = [&](size_t bytes) -> char* {
    off = (off + 255) & ~(size_t)255;
    char* p = base + off; off += bytes; return p;
  };
  const int Etot = N_EDGES + N_NODES;
  half_t* wt16 = (half_t*)carve((size_t)512 * 512 * sizeof(half_t)); // transposed f16 weights
  half_t* x16  = (half_t*)carve((size_t)N_NODES * 512 * sizeof(half_t));
  float*  hbuf = (float*)carve((size_t)N_NODES * 512 * sizeof(float));
  float*  obuf = (float*)carve((size_t)N_NODES * 512 * sizeof(float)); // acc, then BN/ReLU in place
  float*  deg  = (float*)carve((size_t)N_NODES * sizeof(float));
  float*  lattr= (float*)carve((size_t)N_NODES * EDGE_DIM * sizeof(float));
  float*  ve   = (float*)carve((size_t)EDGE_DIM * 4 * sizeof(float));
  float*  aedge= (float*)carve((size_t)Etot * 4 * sizeof(float));
  float*  pbuf = (float*)carve((size_t)Etot * 4 * sizeof(float));
  float*  asrc = (float*)carve((size_t)N_NODES * 4 * sizeof(float));
  float*  adst = (float*)carve((size_t)N_NODES * 4 * sizeof(float));
  float*  mbuf = (float*)carve((size_t)N_NODES * 4 * sizeof(float));
  float*  dbuf = (float*)carve((size_t)N_NODES * 4 * sizeof(float));
  float*  psum = (float*)carve((size_t)N_GRAPH * 32 * sizeof(float));
  float*  pmax = (float*)carve((size_t)N_GRAPH * 32 * sizeof(float));
  float*  pcnt = (float*)carve((size_t)N_GRAPH * sizeof(float));

  auto nb = [](long long n, int bs) { return (unsigned)((n + bs - 1) / bs); };
  const int BS = 256;

  // self-loop attr = per-destination mean of incoming edge_attr
  fill_f32_kernel<<<nb(N_NODES, BS), BS, 0, stream>>>(deg, 0.f, N_NODES);
  fill_f32_kernel<<<nb((long long)N_NODES * EDGE_DIM, BS), BS, 0, stream>>>(lattr, 0.f, N_NODES * EDGE_DIM);
  deg_attr_kernel<<<nb(N_EDGES, BS), BS, 0, stream>>>(dst, eattr, deg, lattr, N_EDGES);
  loop_div_kernel<<<nb(N_NODES, BS), BS, 0, stream>>>(lattr, deg, N_NODES);

  // fp16 copy of input features for WMMA
  cvt_f32_f16_kernel<<<nb((long long)N_NODES * INCH, BS), BS, 0, stream>>>(x, x16, N_NODES * INCH);

  for (int li = 0; li < 3; ++li) {
    const LayerP& P = L[li];
    const int M = P.Hh * P.C;
    // weights -> f16, transposed so B fragments are contiguous per lane
    cvt_transpose_f16_kernel<<<nb((long long)P.Kin * M, BS), BS, 0, stream>>>(P.W, wt16, P.Kin, M);
    // h = x @ W  (WMMA f16 x f16 -> f32)
    dim3 gg(N_NODES / 32, (M + 63) / 64);
    gemm_wmma_f16<<<gg, 128, 0, stream>>>(x16, wt16, hbuf, P.Kin, M);
    // attention coefficients
    attn_node_kernel<<<nb((long long)N_NODES * P.Hh, BS), BS, 0, stream>>>(hbuf, P.as_, P.ad_, asrc, adst, N_NODES, P.Hh, P.C);
    ve_kernel<<<1, 32, 0, stream>>>(P.We, P.ae_, ve, P.Hh, P.C);
    aedge_kernel<<<nb(Etot, BS), BS, 0, stream>>>(eattr, lattr, ve, aedge, P.Hh);
    // softmax over destinations + aggregation
    fill_f32_kernel<<<nb((long long)N_NODES * P.Hh, BS), BS, 0, stream>>>(mbuf, -INFINITY, N_NODES * P.Hh);
    fill_f32_kernel<<<nb((long long)N_NODES * P.Hh, BS), BS, 0, stream>>>(dbuf, 0.f, N_NODES * P.Hh);
    fill_f32_kernel<<<nb((long long)N_NODES * M, BS), BS, 0, stream>>>(obuf, 0.f, N_NODES * M);
    edge_logit_kernel<<<nb((long long)Etot * P.Hh, BS), BS, 0, stream>>>(src, dst, asrc, adst, aedge, pbuf, mbuf, P.Hh);
    edge_exp_kernel<<<nb((long long)Etot * P.Hh, BS), BS, 0, stream>>>(dst, pbuf, mbuf, dbuf, P.Hh);
    edge_msg_kernel<<<nb((long long)Etot * M, BS), BS, 0, stream>>>(src, dst, pbuf, dbuf, hbuf, obuf, P.Hh, P.C);
    // bias + BN + ReLU (in place), f16 copy feeds next layer's GEMM
    post_bn_kernel<<<nb((long long)N_NODES * M, BS), BS, 0, stream>>>(obuf, P.b, P.g, P.bt, P.rm, P.rv, obuf, x16, N_NODES, M);
  }

  // graph pooling (h3 lives in obuf, [N,32]) + MLP head
  fill_f32_kernel<<<nb(N_GRAPH * 32, BS), BS, 0, stream>>>(psum, 0.f, N_GRAPH * 32);
  fill_f32_kernel<<<nb(N_GRAPH * 32, BS), BS, 0, stream>>>(pmax, -INFINITY, N_GRAPH * 32);
  fill_f32_kernel<<<nb(N_GRAPH, BS), BS, 0, stream>>>(pcnt, 0.f, N_GRAPH);
  pool_kernel<<<nb(N_NODES, BS), BS, 0, stream>>>(obuf, batch, psum, pmax, pcnt, N_NODES);
  mlp_kernel<<<1, 64, 0, stream>>>(psum, pmax, pcnt, lw1, lb1, lw2, lb2, outp);
}
